// RelativePositionAttention_2422361555012
// MI455X (gfx1250) — matrix operand order repaired and checked by reading, not compiled, not run on hardware
//
#include <hip/hip_runtime.h>

// Problem constants (from reference)
#define BATCH  4
#define SEQ    1024
#define DMODEL 1024
#define NHEADS 16
#define HD     64       // head dim
#define BQ     128      // queries per workgroup (8 waves x 16)
#define BKV    64       // kv rows staged in LDS per outer step
#define KSTR   72       // padded LDS row stride (halves) for Ksh / VTsh (16B-aligned chunks)
#define PSTR   72       // padded LDS row stride (halves) for P scratch (16x64 tile per wave)

typedef __attribute__((ext_vector_type(16))) _Float16 v16h;
typedef __attribute__((ext_vector_type(8)))  float    v8f;

union H16 { v16h v; float4 f[2]; _Float16 h[16]; };

// Load a v16h WMMA operand as two 16-byte LDS reads (ds_load_b128 x2)
__device__ __forceinline__ v16h lds_ld16(const _Float16* p0, const _Float16* p1) {
  H16 u;
  u.f[0] = *(const float4*)p0;
  u.f[1] = *(const float4*)p1;
  return u.v;
}

__device__ __forceinline__ v8f wmma_f16(v16h a, v16h b, v8f c) {
  // D = A(16x32 f16) * B(32x16 f16) + C(16x16 f32)
  return __builtin_amdgcn_wmma_f32_16x16x32_f16(false, a, false, b, (short)0, c,
                                                false, false);
}

// Raw max without the backend's canonicalize (inputs are finite scores).
__device__ __forceinline__ float fmax_raw(float a, float b) {
  float r;
  asm("v_max_num_f32 %0, %1, %2" : "=v"(r) : "v"(a), "v"(b));
  return r;
}

// xor-butterfly shuffle confined to 16-lane groups, pure VALU via v_permlane16
// (selectors: lane j reads lane j^m).  Fallbacks keep this compiling anywhere.
template <int M, unsigned SEL0, unsigned SEL1>
__device__ __forceinline__ float lane16_xor(float x) {
#if __has_builtin(__builtin_amdgcn_permlane16)
  unsigned u = __float_as_uint(x);
  unsigned r = __builtin_amdgcn_permlane16(u, u, SEL0, SEL1, false, false);
  return __uint_as_float(r);
#elif __has_builtin(__builtin_amdgcn_ds_swizzle)
  // group-of-32 swizzle: and=0x1f, or=0, xor=M  -> imm = (M<<10)|0x1f
  return __int_as_float(
      __builtin_amdgcn_ds_swizzle(__float_as_int(x), (M << 10) | 0x1f));
#else
  return __shfl_xor(x, M, 32);
#endif
}

__device__ __forceinline__ float rowmax16(float x) {
  x = fmax_raw(x, lane16_xor<1, 0x67452301u, 0xEFCDAB89u>(x));
  x = fmax_raw(x, lane16_xor<2, 0x54761032u, 0xDCFE98BAu>(x));
  x = fmax_raw(x, lane16_xor<4, 0x32107654u, 0xBA98FEDCu>(x));
  x = fmax_raw(x, lane16_xor<8, 0xFEDCBA98u, 0x76543210u>(x));
  return x;
}

__global__ __launch_bounds__(256)
void relpos_flash_attn_kernel(const float* __restrict__ query,
                              const float* __restrict__ key,
                              const float* __restrict__ value,
                              const float* __restrict__ rel_k,
                              const float* __restrict__ rel_v,
                              float* __restrict__ out)
{
  // K' = scale*K + rel_k staged row-major [kv][d]; V' = V + rel_v staged
  // transposed [d][kv] so both GEMMs' B operands are contiguous 32B reads.
  __shared__ _Float16 Ksh [BKV * KSTR];
  __shared__ _Float16 VTsh[HD  * KSTR];
  __shared__ _Float16 Psh [8 * 16 * PSTR];   // per-wave P scratch (C/D -> A relayout)

  const int nqb  = SEQ / BQ;
  const int qblk = blockIdx.x % nqb;
  const int bh   = blockIdx.x / nqb;
  const int b    = bh / NHEADS;
  const int h    = bh % NHEADS;

  const int tid  = threadIdx.x;
  const int wid  = tid >> 5;
  const int lane = tid & 31;
  const int hi   = lane >> 4;    // half-wave group (ISA layout split)
  const int lm   = lane & 15;

  const int qw = qblk * BQ + wid * 16;       // first query row of this wave

  // ---- Q in A-matrix layout (loop invariant): lane L -> row M = L&15,
  // halves 0..15 -> K = hi*16+i  (per tile of 32)
  const float* qrow = query + ((size_t)(b * SEQ + qw + lm)) * DMODEL + h * HD;
  H16 a0u, a1u;
#pragma unroll
  for (int i = 0; i < 8; ++i) {
    a0u.h[i]     = (_Float16)qrow[     hi * 16 + i];
    a0u.h[8 + i] = (_Float16)qrow[ 8 + hi * 16 + i];
    a1u.h[i]     = (_Float16)qrow[32 + hi * 16 + i];
    a1u.h[8 + i] = (_Float16)qrow[40 + hi * 16 + i];
  }
  const v16h qa0 = a0u.v, qa1 = a1u.v;

  v16h onesB;
  {
    H16 u;
#pragma unroll
    for (int i = 0; i < 16; ++i) u.h[i] = (_Float16)1.0f;
    onesB = u.v;  // all-ones B operand: WMMA row-sum trick
  }

  const v8f vzero = {0.f, 0.f, 0.f, 0.f, 0.f, 0.f, 0.f, 0.f};
  v8f Oacc[4];
#pragma unroll
  for (int nt = 0; nt < 4; ++nt) Oacc[nt] = vzero;
  float m_r[8], l_r[8];
#pragma unroll
  for (int r = 0; r < 8; ++r) { m_r[r] = -1e30f; l_r[r] = 0.0f; }

  const float scale = 0.125f;  // 1/sqrt(64)

  _Float16*       Pw   = &Psh[wid * 16 * PSTR];
  const _Float16* Prow = &Psh[(wid * 16 + lm) * PSTR];

  // cooperative tile-load mapping: 4 threads per kv row, 16 cols each
  const int crow = tid >> 2;
  const int ccol = (tid & 3) * 16;

  for (int kv0 = 0; kv0 < SEQ; kv0 += BKV) {
    __syncthreads();
    {
      const float* kr = key   + ((size_t)(b * SEQ + kv0 + crow)) * DMODEL + h * HD + ccol;
      const float* rk = rel_k + (size_t)(kv0 + crow) * HD + ccol;
      const float* vr = value + ((size_t)(b * SEQ + kv0 + crow)) * DMODEL + h * HD + ccol;
      const float* rv = rel_v + (size_t)(kv0 + crow) * HD + ccol;
      H16 kk;
#pragma unroll
      for (int i = 0; i < 16; ++i) kk.h[i] = (_Float16)(scale * kr[i] + rk[i]);
      float4* kd = (float4*)&Ksh[crow * KSTR + ccol];
      kd[0] = kk.f[0];
      kd[1] = kk.f[1];
#pragma unroll
      for (int i = 0; i < 16; ++i)
        VTsh[(ccol + i) * KSTR + crow] = (_Float16)(vr[i] + rv[i]);
      if (kv0 + BKV < SEQ) {
        __builtin_prefetch(kr + (size_t)BKV * DMODEL, 0, 3);
        __builtin_prefetch(vr + (size_t)BKV * DMODEL, 0, 3);
      }
    }
    __syncthreads();

    // ---- scores S(16x64) = Q x K'^T : four 16x16 n-tiles, computed up front
    // so the softmax update runs ONCE per kv64 block.  Within each 32-chunk
    // tile pairs are column-interleaved (even kv / odd kv) so each row's
    // (p_even, p_odd) pair packs into one b32 LDS store.
    v8f s0, s1, s2, s3;
    {
      const _Float16* kpe0 = &Ksh[(     2 * lm    ) * KSTR + hi * 16];
      const _Float16* kpo0 = &Ksh[(     2 * lm + 1) * KSTR + hi * 16];
      const _Float16* kpe1 = &Ksh[(32 + 2 * lm    ) * KSTR + hi * 16];
      const _Float16* kpo1 = &Ksh[(32 + 2 * lm + 1) * KSTR + hi * 16];
      s0 = wmma_f16(qa0, lds_ld16(kpe0,      kpe0 + 8),  vzero);
      s0 = wmma_f16(qa1, lds_ld16(kpe0 + 32, kpe0 + 40), s0);
      s1 = wmma_f16(qa0, lds_ld16(kpo0,      kpo0 + 8),  vzero);
      s1 = wmma_f16(qa1, lds_ld16(kpo0 + 32, kpo0 + 40), s1);
      s2 = wmma_f16(qa0, lds_ld16(kpe1,      kpe1 + 8),  vzero);
      s2 = wmma_f16(qa1, lds_ld16(kpe1 + 32, kpe1 + 40), s2);
      s3 = wmma_f16(qa0, lds_ld16(kpo1,      kpo1 + 8),  vzero);
      s3 = wmma_f16(qa1, lds_ld16(kpo1 + 32, kpo1 + 40), s3);
    }

    // ---- single online-softmax update over the 64 columns
    float alpha[8];
#pragma unroll
    for (int r = 0; r < 8; ++r) {
      float mx = fmax_raw(fmax_raw(s0[r], s1[r]), fmax_raw(s2[r], s3[r]));
      mx = rowmax16(mx);
      const float mn = fmax_raw(m_r[r], mx);
      alpha[r] = __expf(m_r[r] - mn);
      m_r[r]   = mn;
      const float p0 = __expf(s0[r] - mn);
      const float p1 = __expf(s1[r] - mn);
      const float p2 = __expf(s2[r] - mn);
      const float p3 = __expf(s3[r] - mn);
      union { _Float16 hh[2]; unsigned u; } pkA, pkB;
      pkA.hh[0] = (_Float16)p0;
      pkA.hh[1] = (_Float16)p1;
      pkB.hh[0] = (_Float16)p2;
      pkB.hh[1] = (_Float16)p3;
      *(unsigned*)&Pw[(r + hi * 8) * PSTR +      2 * lm] = pkA.u;
      *(unsigned*)&Pw[(r + hi * 8) * PSTR + 32 + 2 * lm] = pkB.u;
    }
#pragma unroll
    for (int nt = 0; nt < 4; ++nt)
#pragma unroll
      for (int r = 0; r < 8; ++r) Oacc[nt][r] *= alpha[r];

    // ---- reload P(16x64) as two A operands (LDS is in-order within a wave)
    const v16h pa0 = lds_ld16(Prow +      hi * 16, Prow +  8 + hi * 16);
    const v16h pa1 = lds_ld16(Prow + 32 + hi * 16, Prow + 40 + hi * 16);

    // row sums via ones-B WMMAs; fold the alpha rescale of l into one FMA
    v8f rs = wmma_f16(pa0, onesB, vzero);
    rs     = wmma_f16(pa1, onesB, rs);
#pragma unroll
    for (int r = 0; r < 8; ++r) l_r[r] = fmaf(l_r[r], alpha[r], rs[r]);

    // ---- O += P x V' : 4 n-tiles over d=64, contraction kv=64 in two steps
#pragma unroll
    for (int nt = 0; nt < 4; ++nt) {
      const _Float16* vp = &VTsh[(nt * 16 + lm) * KSTR + hi * 16];
      Oacc[nt] = wmma_f16(pa0, lds_ld16(vp,      vp + 8),  Oacc[nt]);
      Oacc[nt] = wmma_f16(pa1, lds_ld16(vp + 32, vp + 40), Oacc[nt]);
    }
  }

  // ---- normalize and store: out[b][q][h*64 + d]
  float inv[8];
#pragma unroll
  for (int r = 0; r < 8; ++r) inv[r] = 1.0f / l_r[r];
#pragma unroll
  for (int nt = 0; nt < 4; ++nt) {
#pragma unroll
    for (int r = 0; r < 8; ++r) {
      out[((size_t)(b * SEQ + qw + r + hi * 8)) * DMODEL + h * HD + nt * 16 + lm] =
          Oacc[nt][r] * inv[r];
    }
  }
}

extern "C" void kernel_launch(void* const* d_in, const int* in_sizes, int n_in,
                              void* d_out, int out_size, void* d_ws, size_t ws_size,
                              hipStream_t stream) {
  (void)in_sizes; (void)n_in; (void)out_size; (void)d_ws; (void)ws_size;
  const float* q  = (const float*)d_in[0];
  const float* k  = (const float*)d_in[1];
  const float* v  = (const float*)d_in[2];
  const float* rk = (const float*)d_in[3];
  const float* rv = (const float*)d_in[4];
  float* o = (float*)d_out;
  dim3 grid(BATCH * NHEADS * (SEQ / BQ));  // 512 workgroups
  dim3 block(256);                          // 8 wave32 waves
  hipLaunchKernelGGL(relpos_flash_attn_kernel, grid, block, 0, stream,
                     q, k, v, rk, rv, o);
}
